// Attention_63015760167583
// MI455X (gfx1250) — matrix operand order repaired and checked by reading, not compiled, not run on hardware
//
#include <hip/hip_runtime.h>

// ---------------------------------------------------------------------------
// MI455X (gfx1250) attention layer.
// Matmuls via V_WMMA_F32_16X16X32_BF16 (f32 accum); global->LDS staging via
// GLOBAL_LOAD_ASYNC_TO_LDS_B128 (ASYNCcnt), double-buffered in the GEMMs.
// Softmax row reductions via DPP8/DPP16 (no ds_bpermute).
// ---------------------------------------------------------------------------

typedef __attribute__((ext_vector_type(16))) __bf16 v16bf;
typedef __attribute__((ext_vector_type(8)))  float  v8f;

union FragU {
  v16bf          bf;
  unsigned short u[16];
  uint4          q[2];
};

__device__ __forceinline__ unsigned short f2bf(float f) {
  unsigned u = __builtin_bit_cast(unsigned, f);
  u += 0x7FFFu + ((u >> 16) & 1u);           // round-to-nearest-even
  return (unsigned short)(u >> 16);
}
__device__ __forceinline__ float bf2f(unsigned short h) {
  return __builtin_bit_cast(float, ((unsigned)h) << 16);
}

// LDS byte offset of a __shared__ object: low 32 bits of the generic pointer
__device__ __forceinline__ unsigned lds_off(const void* p) {
  return (unsigned)(uintptr_t)p;
}

// 16-byte async DMA global -> LDS (per-lane addresses, tracked by ASYNCcnt).
__device__ __forceinline__ void async_copy16(unsigned lds_addr, const void* gaddr) {
  asm volatile("global_load_async_to_lds_b128 %0, %1, off"
               :: "v"(lds_addr), "v"(gaddr) : "memory");
}
__device__ __forceinline__ void wait_async0() {
  asm volatile("s_wait_asynccnt 0x0" ::: "memory");
}
__device__ __forceinline__ void wait_ds0() {
  asm volatile("s_wait_dscnt 0x0" ::: "memory");
}

// ---- DPP 16-lane butterfly reductions (stay inside each 16-lane half) ------
#define DPP8SEL(a,b,c,d,e,f,g,h) \
  ((a)|((b)<<3)|((c)<<6)|((d)<<9)|((e)<<12)|((f)<<15)|((g)<<18)|((h)<<21))

template <int SEL>
__device__ __forceinline__ float dpp8_f(float x) {
  return __builtin_bit_cast(float,
      __builtin_amdgcn_mov_dpp8(__builtin_bit_cast(int, x), SEL));
}
__device__ __forceinline__ float row_mirror_f(float x) {
  // DPP16 row_mirror (0x140): lane i -> 15-i within each row of 16
  return __builtin_bit_cast(float,
      __builtin_amdgcn_mov_dpp(__builtin_bit_cast(int, x), 0x140, 0xF, 0xF, true));
}
__device__ __forceinline__ float red16_max(float x) {
  x = fmaxf(x, dpp8_f<DPP8SEL(1,0,3,2,5,4,7,6)>(x));   // xor 1
  x = fmaxf(x, dpp8_f<DPP8SEL(2,3,0,1,6,7,4,5)>(x));   // xor 2
  x = fmaxf(x, dpp8_f<DPP8SEL(4,5,6,7,0,1,2,3)>(x));   // xor 4
  x = fmaxf(x, row_mirror_f(x));                       // combine 8-groups
  return x;
}
__device__ __forceinline__ float red16_sum(float x) {
  x += dpp8_f<DPP8SEL(1,0,3,2,5,4,7,6)>(x);
  x += dpp8_f<DPP8SEL(2,3,0,1,6,7,4,5)>(x);
  x += dpp8_f<DPP8SEL(4,5,6,7,0,1,2,3)>(x);
  x += row_mirror_f(x);
  return x;
}

constexpr int Bsz = 4, Sseq = 2048, Dm = 1024, Hh = 16, DH = 64;
constexpr int Mrows = Bsz * Sseq;            // 8192
constexpr float NEGBIG = -1e30f;

// ---------------------------------------------------------------------------
// One-shot f32 -> bf16 conversion (x and the four weights; tiny vs 23.3 TB/s).
// ---------------------------------------------------------------------------
__global__ __launch_bounds__(256) void cvt_f32_bf16(const float* __restrict__ in,
                                                    unsigned short* __restrict__ out,
                                                    int n4) {
  int i = blockIdx.x * 256 + threadIdx.x;
  if (i >= n4) return;
  float4 v = ((const float4*)in)[i];
  ushort4 o;
  o.x = f2bf(v.x); o.y = f2bf(v.y); o.z = f2bf(v.z); o.w = f2bf(v.w);
  ((ushort4*)out)[i] = o;
}

// ---------------------------------------------------------------------------
// GEMM: C[M,N] = A[M,K] * W[N,K]^T, A/W bf16, C bf16 or f32.
// Block tile 128x128x64, double-buffered async-to-LDS staging.
// 8 waves -> 2x4 wave grid, each wave 64x32 => 16 WMMA per k-step.
// ---------------------------------------------------------------------------
template <int OUT_BF16>
__global__ __launch_bounds__(256) void gemm_bf16(const unsigned short* __restrict__ A,
                                                 const unsigned short* __restrict__ W,
                                                 void* __restrict__ Cout,
                                                 int M, int N, int Kd) {
  __shared__ unsigned short As[2][128][64];   // 16 KB each buffer
  __shared__ unsigned short Ws[2][128][64];

  const int t    = threadIdx.x;
  const int lane = t & 31;
  const int half = (lane >= 16) ? 1 : 0;
  const int l16  = lane & 15;
  const int wid  = t >> 5;
  const int m0   = blockIdx.y * 128;
  const int n0   = blockIdx.x * 128;
  const int wm   = wid >> 2;                 // 0..1  (rows, 64 each)
  const int wn   = wid & 3;                  // 0..3  (cols, 32 each)

  v8f acc[4][2];
#pragma unroll
  for (int mi = 0; mi < 4; ++mi)
#pragma unroll
    for (int ni = 0; ni < 2; ++ni) acc[mi][ni] = {};

  // issue async copies for one 128x64 double-tile (A and W): 4+4 B128/thread
  auto issue_tiles = [&](int buf, int kb) {
#pragma unroll
    for (int i = 0; i < 4; ++i) {
      int id = i * 256 + t;                  // 16B chunk id (1024 per tile)
      int r  = id >> 3;
      int c8 = (id & 7) * 8;                 // ushort offset in row
      async_copy16(lds_off(&As[buf][r][c8]), A + (size_t)(m0 + r) * Kd + kb + c8);
      async_copy16(lds_off(&Ws[buf][r][c8]), W + (size_t)(n0 + r) * Kd + kb + c8);
    }
  };

  int buf = 0;
  issue_tiles(0, 0);
  for (int kb = 0; kb < Kd; kb += 64) {
    wait_async0();
    __syncthreads();                         // tiles[buf] visible to all waves
    if (kb + 64 < Kd) issue_tiles(buf ^ 1, kb + 64);

#pragma unroll
    for (int kh = 0; kh < 2; ++kh) {         // two 32-wide k-halves
      FragU a[4], b[2];
#pragma unroll
      for (int mi = 0; mi < 4; ++mi) {
        // A-layout (16-bit, 16x32): M = lane%16; k = (lane<16?0:8)+(i/8)*16+i%8
        const unsigned short* p = &As[buf][wm * 64 + mi * 16 + l16][kh * 32 + half * 8];
        a[mi].q[0] = *(const uint4*)p;
        a[mi].q[1] = *(const uint4*)(p + 16);
      }
#pragma unroll
      for (int ni = 0; ni < 2; ++ni) {
        // B-layout (16-bit, 32x16): N = lane%16; k = (lane<16?0:8)+(i/8)*16+i%8
        const unsigned short* p = &Ws[buf][wn * 32 + ni * 16 + l16][kh * 32 + half * 8];
        b[ni].q[0] = *(const uint4*)p;
        b[ni].q[1] = *(const uint4*)(p + 16);
      }
#pragma unroll
      for (int mi = 0; mi < 4; ++mi)
#pragma unroll
        for (int ni = 0; ni < 2; ++ni)
          acc[mi][ni] = __builtin_amdgcn_wmma_f32_16x16x32_bf16(
              false, a[mi].bf, false, b[ni].bf, (short)0, acc[mi][ni], false, false);
    }
    buf ^= 1;
  }

  // C-layout: VGPR r -> M = r + 8*half, N = lane%16
#pragma unroll
  for (int mi = 0; mi < 4; ++mi)
#pragma unroll
    for (int ni = 0; ni < 2; ++ni)
#pragma unroll
      for (int r = 0; r < 8; ++r) {
        int row = m0 + wm * 64 + mi * 16 + r + half * 8;
        int col = n0 + wn * 32 + ni * 16 + l16;
        float v = acc[mi][ni][r];
        if (OUT_BF16)
          ((unsigned short*)Cout)[(size_t)row * N + col] = f2bf(v);
        else
          ((float*)Cout)[(size_t)row * N + col] = v;
      }
}

// ---------------------------------------------------------------------------
// Interleaved RoPE on a bf16 (Mrows x Dm) buffer, in place.
// skip_phase_rope: rows with s < E untouched, else position = s - E.
// ---------------------------------------------------------------------------
__global__ __launch_bounds__(256) void rope_kernel(unsigned short* T,
                                                   const int* pE,
                                                   const int* pSkip) {
  int idx = blockIdx.x * 256 + threadIdx.x;  // pair id, Mrows*Dm/2 total
  int row = idx >> 9;                        // / (Dm/2)
  int pr  = idx & 511;
  int s   = row & (Sseq - 1);
  int E = *pE, skip = *pSkip;
  int pos;
  if (skip) { if (s < E) return; pos = s - E; } else { pos = s; }
  int hp = pr & (DH / 2 - 1);                // pair index within head
  float inv = exp2f(-(float)(2 * hp) * (13.287712379549449f / 64.0f));
  float ang = (float)pos * inv;
  float sn, cs;
  __sincosf(ang, &sn, &cs);
  unsigned short* p = T + ((size_t)row << 10) + (pr << 1);
  float x0 = bf2f(p[0]), x1 = bf2f(p[1]);
  p[0] = f2bf(x0 * cs - x1 * sn);
  p[1] = f2bf(x1 * cs + x0 * sn);
}

// ---------------------------------------------------------------------------
// Flash attention. Grid: (Sseq/128, Bsz*Hh), 256 threads (8 waves).
// Each wave: 16 query rows, kv blocks of 32. mask = (j<=i) | (j<E), + amask.
// V staged via async-to-LDS, overlapped with QK^T WMMA + softmax.
// ---------------------------------------------------------------------------
__global__ __launch_bounds__(256) void attn_kernel(const unsigned short* __restrict__ Q,
                                                   const unsigned short* __restrict__ K,
                                                   const unsigned short* __restrict__ V,
                                                   const unsigned char* __restrict__ amask,
                                                   const int* pE,
                                                   unsigned short* __restrict__ AO) {
  __shared__ unsigned short Plds[8][16 * 32];     // per-wave P (16x32) bf16
  __shared__ unsigned short Vlds[8][32][64];      // per-wave V block (32x64)

  const int lane = threadIdx.x & 31;
  const int wid  = threadIdx.x >> 5;
  const int half = (lane >= 16) ? 1 : 0;
  const int l16  = lane & 15;
  const int b    = blockIdx.y >> 4;
  const int hd   = blockIdx.y & 15;
  const int row_base = blockIdx.x * 128 + wid * 16;
  const int E    = *pE;

  // Q fragments (scale 1/sqrt(64) folded in), one per 32-wide k-half of dh
  FragU qa[2];
  {
    const unsigned short* qrow =
        Q + ((size_t)(b * Sseq + row_base + l16)) * Dm + hd * DH;
#pragma unroll
    for (int kh = 0; kh < 2; ++kh) {
      int base = kh * 32 + half * 8;
#pragma unroll
      for (int i = 0; i < 16; ++i) {
        int k = base + (i >> 3) * 16 + (i & 7);
        qa[kh].u[i] = f2bf(bf2f(qrow[k]) * 0.125f);
      }
    }
  }

  v8f o[4];
#pragma unroll
  for (int nt = 0; nt < 4; ++nt) o[nt] = {};
  float m[8], l[8];
#pragma unroll
  for (int r = 0; r < 8; ++r) { m[r] = NEGBIG; l[r] = 0.0f; }

  int jend = row_base + 16;                  // causal horizon for this wave
  if (E > jend) jend = E;                    // phase block always visible
  if (jend > Sseq) jend = Sseq;
  const int nb = (jend + 31) >> 5;

  for (int jb = 0; jb < nb; ++jb) {
    const int j0 = jb * 32;

    // async-stage V block (32 rows x 64) into this wave's LDS region.
    // wait_ds0 first: prior-iteration Vlds reads must land before overwrite.
    wait_ds0();
    {
      const unsigned short* vrow =
          V + ((size_t)(b * Sseq + j0 + lane)) * Dm + hd * DH;
      unsigned lbase = lds_off(&Vlds[wid][lane][0]);
#pragma unroll
      for (int g = 0; g < 8; ++g)
        async_copy16(lbase + g * 16, vrow + g * 8);
    }

    // scores: S = Q * K^T  (two 16x16 n-tiles, accumulate over dh=64)
    v8f s0 = {}, s1 = {};
#pragma unroll
    for (int kh = 0; kh < 2; ++kh) {
      FragU kb0, kb1;
      const unsigned short* kr0 =
          K + ((size_t)(b * Sseq + j0 + l16)) * Dm + hd * DH + kh * 32 + half * 8;
      const unsigned short* kr1 = kr0 + (size_t)16 * Dm;
      kb0.q[0] = *(const uint4*)kr0; kb0.q[1] = *(const uint4*)(kr0 + 16);
      kb1.q[0] = *(const uint4*)kr1; kb1.q[1] = *(const uint4*)(kr1 + 16);
      s0 = __builtin_amdgcn_wmma_f32_16x16x32_bf16(false, qa[kh].bf, false, kb0.bf,
                                                   (short)0, s0, false, false);
      s1 = __builtin_amdgcn_wmma_f32_16x16x32_bf16(false, qa[kh].bf, false, kb1.bf,
                                                   (short)0, s1, false, false);
    }

    // additive masking: visible -> +0, masked -> <= -1e30 (exp underflows to 0)
    const int col0 = j0 + l16, col1 = col0 + 16;
    const float amb0 = (amask[(size_t)b * Sseq + col0] != 0) ? 0.0f : NEGBIG;
    const float amb1 = (amask[(size_t)b * Sseq + col1] != 0) ? 0.0f : NEGBIG;
    const float phb0 = (col0 < E) ? 0.0f : NEGBIG;
    const float phb1 = (col1 < E) ? 0.0f : NEGBIG;

#pragma unroll
    for (int r = 0; r < 8; ++r) {
      int row = row_base + r + half * 8;
      float cb0 = (col0 <= row) ? 0.0f : NEGBIG;
      float cb1 = (col1 <= row) ? 0.0f : NEGBIG;
      float a0 = s0[r] + amb0 + fmaxf(cb0, phb0);
      float a1 = s1[r] + amb1 + fmaxf(cb1, phb1);
      float mx = red16_max(fmaxf(a0, a1));
      float mn = fmaxf(m[r], mx);
      float al = __expf(m[r] - mn);
      float p0 = __expf(a0 - mn);
      float p1 = __expf(a1 - mn);
      float sum = red16_sum(p0 + p1);
      m[r] = mn;
      l[r] = l[r] * al + sum;
#pragma unroll
      for (int nt = 0; nt < 4; ++nt) o[nt][r] *= al;
      int prow = r + half * 8;
      Plds[wid][prow * 32 + l16]      = f2bf(p0);
      Plds[wid][prow * 32 + 16 + l16] = f2bf(p1);
    }

    // wave-private LDS transpose: wait for our own DS stores before re-reading
    wait_ds0();

    // P as A-fragment (16x32)
    FragU pa;
    {
      const unsigned short* pp = &Plds[wid][l16 * 32 + half * 8];
      pa.q[0] = *(const uint4*)pp;
      pa.q[1] = *(const uint4*)(pp + 16);
    }

    // V block ready (async DMA from block top)
    wait_async0();

    // O += P * V  (4 n-tiles over dh)
#pragma unroll
    for (int nt = 0; nt < 4; ++nt) {
      FragU vb;
#pragma unroll
      for (int i = 0; i < 16; ++i)
        vb.u[i] = Vlds[wid][half * 8 + (i >> 3) * 16 + (i & 7)][nt * 16 + l16];
      o[nt] = __builtin_amdgcn_wmma_f32_16x16x32_bf16(false, pa.bf, false, vb.bf,
                                                      (short)0, o[nt], false, false);
    }
  }

  // normalize and write (B, S, h*dh) in bf16 (feeds the bf16 Wo GEMM)
#pragma unroll
  for (int r = 0; r < 8; ++r) {
    float rl = 1.0f / l[r];
    int row = row_base + r + half * 8;
#pragma unroll
    for (int nt = 0; nt < 4; ++nt) {
      int col = hd * DH + nt * 16 + l16;
      AO[(size_t)(b * Sseq + row) * Dm + col] = f2bf(o[nt][r] * rl);
    }
  }
}

// ---------------------------------------------------------------------------
extern "C" void kernel_launch(void* const* d_in, const int* in_sizes, int n_in,
                              void* d_out, int out_size, void* d_ws, size_t ws_size,
                              hipStream_t stream) {
  const float* x  = (const float*)d_in[0];
  const float* Wq = (const float*)d_in[1];
  const float* Wk = (const float*)d_in[2];
  const float* Wv = (const float*)d_in[3];
  const float* Wo = (const float*)d_in[4];
  const unsigned char* amask = (const unsigned char*)d_in[5];  // bool (B,S)
  const int* pE    = (const int*)d_in[6];
  const int* pSkip = (const int*)d_in[7];
  float* out = (float*)d_out;

  // workspace (all bf16 except final output):
  //   xb 16MB | Wqb,Wkb,Wvb,Wob 2MB each | Qb,Kb,Vb 16MB each | AOb 16MB
  char* ws = (char*)d_ws;
  const size_t xb_bytes = (size_t)Mrows * Dm * sizeof(unsigned short);   // 16MB
  const size_t wb_bytes = (size_t)Dm * Dm * sizeof(unsigned short);      // 2MB
  unsigned short* xb  = (unsigned short*)(ws);
  unsigned short* Wqb = (unsigned short*)(ws + xb_bytes);
  unsigned short* Wkb = (unsigned short*)(ws + xb_bytes + 1 * wb_bytes);
  unsigned short* Wvb = (unsigned short*)(ws + xb_bytes + 2 * wb_bytes);
  unsigned short* Wob = (unsigned short*)(ws + xb_bytes + 3 * wb_bytes);
  unsigned short* Qb  = (unsigned short*)(ws + xb_bytes + 4 * wb_bytes);
  unsigned short* Kb  = (unsigned short*)(ws + 2 * xb_bytes + 4 * wb_bytes);
  unsigned short* Vb  = (unsigned short*)(ws + 3 * xb_bytes + 4 * wb_bytes);
  unsigned short* AOb = (unsigned short*)(ws + 4 * xb_bytes + 4 * wb_bytes);

  // f32 -> bf16 conversion passes
  cvt_f32_bf16<<<(Mrows * Dm / 4) / 256, 256, 0, stream>>>(x, xb, Mrows * Dm / 4);
  cvt_f32_bf16<<<(Dm * Dm / 4) / 256, 256, 0, stream>>>(Wq, Wqb, Dm * Dm / 4);
  cvt_f32_bf16<<<(Dm * Dm / 4) / 256, 256, 0, stream>>>(Wk, Wkb, Dm * Dm / 4);
  cvt_f32_bf16<<<(Dm * Dm / 4) / 256, 256, 0, stream>>>(Wv, Wvb, Dm * Dm / 4);
  cvt_f32_bf16<<<(Dm * Dm / 4) / 256, 256, 0, stream>>>(Wo, Wob, Dm * Dm / 4);

  dim3 ggemm(Dm / 128, Mrows / 128);  // (8, 64)
  gemm_bf16<1><<<ggemm, 256, 0, stream>>>(xb, Wqb, Qb, Mrows, Dm, Dm);
  gemm_bf16<1><<<ggemm, 256, 0, stream>>>(xb, Wkb, Kb, Mrows, Dm, Dm);
  gemm_bf16<1><<<ggemm, 256, 0, stream>>>(xb, Wvb, Vb, Mrows, Dm, Dm);

  int rope_blocks = (Mrows * (Dm / 2)) / 256;  // 16384
  rope_kernel<<<rope_blocks, 256, 0, stream>>>(Qb, pE, pSkip);
  rope_kernel<<<rope_blocks, 256, 0, stream>>>(Kb, pE, pSkip);

  attn_kernel<<<dim3(Sseq / 128, Bsz * Hh), 256, 0, stream>>>(Qb, Kb, Vb, amask, pE, AOb);

  gemm_bf16<0><<<ggemm, 256, 0, stream>>>(AOb, Wob, out, Mrows, Dm, Dm);
}
